// PC_graph_zwol_PYG_15281493639397
// MI455X (gfx1250) — matrix operand order repaired and checked by reading, not compiled, not run on hardware
//
#include <hip/hip_runtime.h>
#include <hip/hip_bf16.h>

// ---------------- problem constants (from reference setup) ----------------
#define B_     512
#define N_     2048
#define NSENS  784          // sensory nodes: err[:, :784] = 0
#define UPDEND 2038         // update_mask = [784, 2038)
#define LRX    0.5f
#define TSTEPS 8
#define BN     (B_ * N_)    // 1048576

#define CHUNK    64          // K elements staged per LDS chunk
#define LDSTRIDE 144         // bytes per 64-col bf16 row (128 + 16 pad, 16B aligned)
#define ABUFSZ   (64 * LDSTRIDE)

typedef __attribute__((ext_vector_type(16))) __bf16 v16bf;
typedef __attribute__((ext_vector_type(8)))  __bf16 v8bf;
typedef __attribute__((ext_vector_type(8)))  float  v8f;

static __device__ inline __bf16 f2bf(float f) {
    union { float f; unsigned u; } v; v.f = f;
    unsigned r = v.u + 0x7FFFu + ((v.u >> 16) & 1u);   // round-to-nearest-even
    unsigned short h = (unsigned short)(r >> 16);
    union { unsigned short s; __bf16 b; } o; o.s = h;
    return o.b;
}

static __device__ inline v16bf cat8(v8bf lo, v8bf hi) {
    return __builtin_shufflevector(lo, hi, 0,1,2,3,4,5,6,7,8,9,10,11,12,13,14,15);
}

static __device__ inline v8f wmma_bf16(v16bf a, v16bf b, v8f c) {
    // 8 args: (neg_a, A, neg_b, B, c_mod, C, reuse_a, reuse_b)
    return __builtin_amdgcn_wmma_f32_16x16x32_bf16(false, a, false, b, (short)0, c, false, false);
}

// Async-copy one 64-row x 64-col bf16 chunk (8 KB) from global to LDS.
// 128 threads x 4 x 16B segments, GV-mode, tracked with ASYNCcnt.
static __device__ __forceinline__ void async_chunk(const __bf16* __restrict__ src,
                                                   int kk, char* ldsbase, int tid) {
    #pragma unroll
    for (int j = 0; j < 4; ++j) {
        const int s   = j * 128 + tid;       // 512 x 16B segments, coalesced
        const int row = s >> 3;
        const int seg = s & 7;
        const void* g = src + row * N_ + kk + seg * 8;
        unsigned l = (unsigned)(unsigned long long)(ldsbase + row * LDSTRIDE + seg * 16);
        asm volatile("global_load_async_to_lds_b128 %0, %1, off"
                     :: "v"(l), "v"(g) : "memory");
    }
}

static __device__ __forceinline__ void wait_async0() {
    asm volatile("s_wait_asynccnt 0x0" ::: "memory");
}

// ------------------------------------------------------------------
// Kernel 0: w (f32, row-major) -> w_bf16 (row-major) + wT_bf16 (transposed)
// ------------------------------------------------------------------
__global__ __launch_bounds__(256) void prep_w(const float* __restrict__ w,
                                              __bf16* __restrict__ wb,
                                              __bf16* __restrict__ wtb) {
    __shared__ float tile[32][33];
    const int bx = blockIdx.x * 32;
    const int by = blockIdx.y * 32;
    #pragma unroll
    for (int i = threadIdx.y; i < 32; i += 8) {
        float v = w[(by + i) * N_ + bx + threadIdx.x];
        wb[(by + i) * N_ + bx + threadIdx.x] = f2bf(v);
        tile[i][threadIdx.x] = v;
    }
    __syncthreads();
    #pragma unroll
    for (int i = threadIdx.y; i < 32; i += 8) {
        wtb[(bx + i) * N_ + by + threadIdx.x] = f2bf(tile[threadIdx.x][i]);
    }
}

// ------------------------------------------------------------------
// Kernel 1: fx = tanh(vals) -> bf16
// ------------------------------------------------------------------
__global__ __launch_bounds__(256) void tanh_cvt(const float* __restrict__ vals,
                                                __bf16* __restrict__ fx) {
    int i = blockIdx.x * 256 + threadIdx.x;
    if (i < BN) fx[i] = f2bf(tanhf(vals[i]));
}

// ------------------------------------------------------------------
// Kernel 2: pred = fx @ w^T; err = vals - pred; err[:, :NSENS] = 0.
// Block = 4 waves sharing 64 A-rows (m0); each wave owns TWO 16-col N tiles
// (block covers 64x128). A staged in LDS via double-buffered async copies;
// B read direct (L2-resident). Grid (B/64, N/128) = (8, 16), block (32,4).
// ------------------------------------------------------------------
__global__ __launch_bounds__(128) void gemm1_err(const __bf16* __restrict__ fx,
                                                 const __bf16* __restrict__ wb,
                                                 const float* __restrict__ vals,
                                                 float* __restrict__ errf,
                                                 __bf16* __restrict__ errb) {
    __shared__ char smem[2 * ABUFSZ];
    const int tid  = threadIdx.y * 32 + threadIdx.x;
    const int lane = threadIdx.x;
    const int half = lane >> 4;          // 0: lanes 0-15, 1: lanes 16-31
    const int lm   = lane & 15;
    const int m0   = blockIdx.x * 64;
    const int n0   = (blockIdx.y * 8 + threadIdx.y * 2) * 16;
    const int ncol0 = n0 + lm;
    const int ncol1 = n0 + 16 + lm;

    v8f acc[2][4] = {{v8f{}, v8f{}, v8f{}, v8f{}}, {v8f{}, v8f{}, v8f{}, v8f{}}};
    const __bf16* __restrict__ wrow0 = wb + ncol0 * N_;
    const __bf16* __restrict__ wrow1 = wb + ncol1 * N_;
    const __bf16* __restrict__ abase = fx + m0 * N_;

    async_chunk(abase, 0, &smem[0], tid);                     // prologue -> buf0

    for (int kk = 0, it = 0; kk < N_; kk += CHUNK, ++it) {
        wait_async0();                                        // my buf[cur] done
        __syncthreads();                                      // everyone's done
        const int cur = it & 1;
        if (kk + CHUNK < N_)
            async_chunk(abase, kk + CHUNK, &smem[(cur ^ 1) * ABUFSZ], tid);
        const char* abuf = &smem[cur * ABUFSZ];
        #pragma unroll
        for (int ks = 0; ks < 2; ++ks) {                      // two K=32 steps
            const int kb = kk + ks * 32 + half * 16;          // B: 16 contiguous K
            __builtin_prefetch(wrow0 + kb + 256, 0, 0);
            v16bf bfrag0 = cat8(*(const v8bf*)(wrow0 + kb),
                                *(const v8bf*)(wrow0 + kb + 8));
            v16bf bfrag1 = cat8(*(const v8bf*)(wrow1 + kb),
                                *(const v8bf*)(wrow1 + kb + 8));
            #pragma unroll
            for (int t = 0; t < 4; ++t) {                     // A: K {0-15}/{16-31}
                const char* ap = abuf + (t * 16 + lm) * LDSTRIDE + ks * 64 + half * 32;
                v16bf afrag = cat8(*(const v8bf*)ap, *(const v8bf*)(ap + 16));
                acc[0][t] = wmma_bf16(afrag, bfrag0, acc[0][t]);
                acc[1][t] = wmma_bf16(afrag, bfrag1, acc[1][t]);
            }
        }
    }
    // Epilogue: VGPR r holds M = r + 8*half, N = lane%16
    #pragma unroll
    for (int u = 0; u < 2; ++u) {
        const int ncol = (u == 0) ? ncol0 : ncol1;
        #pragma unroll
        for (int t = 0; t < 4; ++t) {
            #pragma unroll
            for (int r = 0; r < 8; ++r) {
                const int m = m0 + t * 16 + half * 8 + r;
                const int idx = m * N_ + ncol;
                float e = (ncol < NSENS) ? 0.0f : (vals[idx] - acc[u][t][r]);
                errf[idx] = e;
                errb[idx] = f2bf(e);
            }
        }
    }
}

// ------------------------------------------------------------------
// Kernel 3: t = err @ w (update columns); vals -= 0.5*(err - (1-tanh^2)*t).
// Same structure: 4 waves share 64 A-rows, two N tiles per wave in [784,2048).
// Grid (B/64, ceil(79/8)) = (8, 10); overflow tiles clamped (dup: cols 2032-2037 updated twice).
// ------------------------------------------------------------------
__global__ __launch_bounds__(128) void gemm2_upd(const __bf16* __restrict__ errb,
                                                 const __bf16* __restrict__ wtb,
                                                 const float* __restrict__ errf,
                                                 float* __restrict__ vals) {
    __shared__ char smem[2 * ABUFSZ];
    const int tid  = threadIdx.y * 32 + threadIdx.x;
    const int lane = threadIdx.x;
    const int half = lane >> 4;
    const int lm   = lane & 15;
    const int m0   = blockIdx.x * 64;
    int nt0 = blockIdx.y * 8 + threadIdx.y * 2;
    int nt1 = nt0 + 1;
    if (nt0 > 78) nt0 = 78;                        // clamp: duplicate tile, same data
    if (nt1 > 78) nt1 = 78;
    const int ncol0 = NSENS + nt0 * 16 + lm;       // in [784, 2048)
    const int ncol1 = NSENS + nt1 * 16 + lm;

    v8f acc[2][4] = {{v8f{}, v8f{}, v8f{}, v8f{}}, {v8f{}, v8f{}, v8f{}, v8f{}}};
    const __bf16* __restrict__ wrow0 = wtb + ncol0 * N_;   // B(k,n)=w[k][n]=wT[n][k]
    const __bf16* __restrict__ wrow1 = wtb + ncol1 * N_;
    const __bf16* __restrict__ abase = errb + m0 * N_;

    async_chunk(abase, 0, &smem[0], tid);

    for (int kk = 0, it = 0; kk < N_; kk += CHUNK, ++it) {
        wait_async0();
        __syncthreads();
        const int cur = it & 1;
        if (kk + CHUNK < N_)
            async_chunk(abase, kk + CHUNK, &smem[(cur ^ 1) * ABUFSZ], tid);
        const char* abuf = &smem[cur * ABUFSZ];
        #pragma unroll
        for (int ks = 0; ks < 2; ++ks) {
            const int kb = kk + ks * 32 + half * 16;
            __builtin_prefetch(wrow0 + kb + 256, 0, 0);
            v16bf bfrag0 = cat8(*(const v8bf*)(wrow0 + kb),
                                *(const v8bf*)(wrow0 + kb + 8));
            v16bf bfrag1 = cat8(*(const v8bf*)(wrow1 + kb),
                                *(const v8bf*)(wrow1 + kb + 8));
            #pragma unroll
            for (int t = 0; t < 4; ++t) {
                const char* ap = abuf + (t * 16 + lm) * LDSTRIDE + ks * 64 + half * 32;
                v16bf afrag = cat8(*(const v8bf*)ap, *(const v8bf*)(ap + 16));
                acc[0][t] = wmma_bf16(afrag, bfrag0, acc[0][t]);
                acc[1][t] = wmma_bf16(afrag, bfrag1, acc[1][t]);
            }
        }
    }
    #pragma unroll
    for (int u = 0; u < 2; ++u) {
        const int ncol = (u == 0) ? ncol0 : ncol1;
        if (ncol < UPDEND) {
            #pragma unroll
            for (int t = 0; t < 4; ++t) {
                #pragma unroll
                for (int r = 0; r < 8; ++r) {
                    const int m = m0 + t * 16 + half * 8 + r;
                    const int idx = m * N_ + ncol;
                    float v    = vals[idx];
                    float fv   = tanhf(v);
                    float dEdx = errf[idx] - (1.0f - fv * fv) * acc[u][t][r];
                    vals[idx]  = v - LRX * dEdx;
                }
            }
        }
    }
}

// ------------------------------------------------------------------
extern "C" void kernel_launch(void* const* d_in, const int* in_sizes, int n_in,
                              void* d_out, int out_size, void* d_ws, size_t ws_size,
                              hipStream_t stream) {
    const float* values = (const float*)d_in[0];
    const float* w      = (const float*)d_in[1];
    float* vals = (float*)d_out;                      // working state lives in d_out

    char* ws = (char*)d_ws;
    const size_t WB = (size_t)N_ * N_ * sizeof(__bf16);     // 8 MB each
    __bf16* w_bf   = (__bf16*)(ws);                         // [0, 8MB)
    __bf16* wt_bf  = (__bf16*)(ws + WB);                    // [8, 16MB)
    __bf16* fx_bf  = (__bf16*)(ws + 2 * WB);                // [16, 18MB)
    __bf16* err_bf = (__bf16*)(ws + 2 * WB + (size_t)BN * 2);   // [18, 20MB)
    float*  err_f  = (float*)(ws + 2 * WB + (size_t)BN * 4);    // [20, 24MB)

    hipMemcpyAsync(vals, values, (size_t)BN * sizeof(float),
                   hipMemcpyDeviceToDevice, stream);

    prep_w<<<dim3(N_ / 32, N_ / 32), dim3(32, 8), 0, stream>>>(w, w_bf, wt_bf);

    for (int t = 0; t < TSTEPS; ++t) {
        tanh_cvt<<<dim3(BN / 256), dim3(256), 0, stream>>>(vals, fx_bf);
        gemm1_err<<<dim3(B_ / 64, N_ / 128), dim3(32, 4), 0, stream>>>(
            fx_bf, w_bf, vals, err_f, err_bf);
        gemm2_upd<<<dim3(B_ / 64, 10), dim3(32, 4), 0, stream>>>(
            err_bf, wt_bf, err_f, vals);
    }
}
